// LISTA_66477503808200
// MI455X (gfx1250) — matrix operand order repaired and checked by reading, not compiled, not run on hardware
//
#include <hip/hip_runtime.h>

typedef __attribute__((ext_vector_type(16))) _Float16 v16h;
typedef __attribute__((ext_vector_type(8)))  float    v8f;

#define IMG_H 512
#define IMG_W 512
#define NIMG  32
#define NPIX  (IMG_H * IMG_W)
#define NTOT  (NIMG * NPIX)   // 8388608
#define NBLK_CONV 2048        // 4 x 16 x 32 conv blocks

// ---------------------------------------------------------------------------
// B fragment: banded weight matrix B[k][n] = w[dy][n-k+2] for 0<=n-k+2<5.
// Layout (16-bit B, 32x16): lanes 0-15 hold N=lane, K=0..7 in e0..7, K=16..23 in e8..15;
// lanes 16-31 hold N=lane-16, K=8..15 in e0..7, K=24..31 in e8..15 (same as A).
// ---------------------------------------------------------------------------
__device__ __forceinline__ v16h build_bfrag(float w0, float w1, float w2,
                                            float w3, float w4, int n, int half) {
  v16h b;
#pragma unroll
  for (int e = 0; e < 16; ++e) {
    int K = (e < 8) ? (half * 8 + e) : (8 + half * 8 + e);
    int idx = n - K + 2;
    float v = 0.0f;
    v = (idx == 0) ? w0 : v;
    v = (idx == 1) ? w1 : v;
    v = (idx == 2) ? w2 : v;
    v = (idx == 3) ? w3 : v;
    v = (idx == 4) ? w4 : v;
    b[e] = (_Float16)v;
  }
  return b;
}

// ---------------------------------------------------------------------------
// A fragment: A[m][k] = patch_row[co + k].  Layout (16-bit A, 16x32):
// lanes 0-15: M=lane, K=0..7 in e0..7, K=16..23 in e8..15;
// lanes 16-31: M=lane-16, K=8..15 in e0..7, K=24..31 in e8..15.
// srow points at LDS row (m + dy); K slots >=20 multiply a zero B column.
// ---------------------------------------------------------------------------
__device__ __forceinline__ v16h build_afrag(const _Float16* srow, int co, int half) {
  v16h a;
#pragma unroll
  for (int e = 0; e < 16; ++e) {
    int K = (e < 8) ? (half * 8 + e) : (8 + half * 8 + e);
    a[e] = srow[co + K];
  }
  return a;
}

// ---------------------------------------------------------------------------
// One LISTA layer: out = relu(conv5x5(x,Wx)*inv_s + bx + conv5x5(y,Wy) + by)
// grid (4, 16, 32): x = 128-px column strip, y = 32-px tile row, z = image.
// block 256 = 8 waves; wave w computes TWO stacked 16x16 tiles at col w*16.
// If do_sum != 0, each block writes a deterministic partial sum of its
// outputs to part[flat_block] (fused global-sum, saves a full tensor pass).
// ---------------------------------------------------------------------------
__global__ __launch_bounds__(256)
void conv_layer(const float* __restrict__ xin, const float* __restrict__ yin,
                const float* __restrict__ Wxl, const float* __restrict__ Bxl,
                const float* __restrict__ Wyl, const float* __restrict__ Byl,
                const float* __restrict__ scale, float* __restrict__ out,
                float* __restrict__ part, int do_sum) {
  __shared__ _Float16 sx[36][144];
  __shared__ _Float16 sy[36][144];
  __shared__ float ssum[256];

  const int tid = threadIdx.x;
  const int bcx = blockIdx.x;     // 0..3
  const int bry = blockIdx.y;     // 0..15
  const int img = blockIdx.z;     // 0..31
  const float* xi = xin + (size_t)img * NPIX;
  const float* yi = yin + (size_t)img * NPIX;
  const int row_base = bry * 32 - 2;
  const int col_base = bcx * 128 - 2;

  // Cooperative, coalesced load of 36x144 halo patch (zero padded) as f16.
  for (int idx = tid; idx < 36 * 144; idx += 256) {
    int r = idx / 144;
    int c = idx - r * 144;
    int gr = row_base + r;
    int gc = col_base + c;
    bool ok = (gr >= 0) && (gr < IMG_H) && (gc >= 0) && (gc < IMG_W);
    float xv = ok ? xi[gr * IMG_W + gc] : 0.0f;
    float yv = ok ? yi[gr * IMG_W + gc] : 0.0f;
    sx[r][c] = (_Float16)xv;
    sy[r][c] = (_Float16)yv;
  }
  __syncthreads();

  const int wave = tid >> 5;
  const int lane = tid & 31;
  const int n    = lane & 15;   // A row M / B-C column N
  const int half = lane >> 4;
  const int co   = wave * 16;   // tile column offset in strip

  v8f accx0 = {}, accy0 = {};   // upper 16x16 tile
  v8f accx1 = {}, accy1 = {};   // lower 16x16 tile
#pragma unroll
  for (int dy = 0; dy < 5; ++dy) {
    v16h bxf = build_bfrag(Wxl[5 * dy + 0], Wxl[5 * dy + 1], Wxl[5 * dy + 2],
                           Wxl[5 * dy + 3], Wxl[5 * dy + 4], n, half);
    v16h byf = build_bfrag(Wyl[5 * dy + 0], Wyl[5 * dy + 1], Wyl[5 * dy + 2],
                           Wyl[5 * dy + 3], Wyl[5 * dy + 4], n, half);
    v16h ax0 = build_afrag(&sx[n + dy][0], co, half);
    accx0 = __builtin_amdgcn_wmma_f32_16x16x32_f16(false, ax0, false, bxf,
                                                   (short)0, accx0, false, false);
    v16h ay0 = build_afrag(&sy[n + dy][0], co, half);
    accy0 = __builtin_amdgcn_wmma_f32_16x16x32_f16(false, ay0, false, byf,
                                                   (short)0, accy0, false, false);
    v16h ax1 = build_afrag(&sx[16 + n + dy][0], co, half);
    accx1 = __builtin_amdgcn_wmma_f32_16x16x32_f16(false, ax1, false, bxf,
                                                   (short)0, accx1, false, false);
    v16h ay1 = build_afrag(&sy[16 + n + dy][0], co, half);
    accy1 = __builtin_amdgcn_wmma_f32_16x16x32_f16(false, ay1, false, byf,
                                                   (short)0, accy1, false, false);
  }

  const float s  = scale[0];
  const float b0 = Bxl[0];
  const float b1 = Byl[0];
  // C/D layout: lane holds col N=n; VGPR i holds row M = half*8 + i.
  float* op = out + (size_t)img * NPIX + (bry * 32) * IMG_W + bcx * 128 + co + n;
  float local = 0.0f;
#pragma unroll
  for (int i = 0; i < 8; ++i) {
    int m = half * 8 + i;
    float o0 = fmaxf(accx0[i] * s + b0 + accy0[i] + b1, 0.0f);
    float o1 = fmaxf(accx1[i] * s + b0 + accy1[i] + b1, 0.0f);
    op[m * IMG_W] = o0;
    op[(16 + m) * IMG_W] = o1;
    local += o0 + o1;
  }

  if (do_sum) {
    __syncthreads();            // patch LDS no longer needed
    ssum[tid] = local;
    __syncthreads();
    for (int st = 128; st > 0; st >>= 1) {
      if (tid < st) ssum[tid] += ssum[tid + st];
      __syncthreads();
    }
    if (tid == 0) {
      int flat = (img * 16 + bry) * 4 + bcx;   // 0..2047
      part[flat] = ssum[0];
    }
  }
}

// ------------------------------- reductions --------------------------------
__global__ __launch_bounds__(256)
void minmax_partial(const float* __restrict__ x, float* __restrict__ part) {
  __shared__ float smn[256], smx[256];
  int t = threadIdx.x;
  float mn = 3.402823466e38f, mx = -3.402823466e38f;
  for (int i = blockIdx.x * 256 + t; i < NTOT; i += gridDim.x * 256) {
    float v = x[i];
    mn = fminf(mn, v);
    mx = fmaxf(mx, v);
  }
  smn[t] = mn; smx[t] = mx;
  __syncthreads();
  for (int s = 128; s > 0; s >>= 1) {
    if (t < s) { smn[t] = fminf(smn[t], smn[t + s]); smx[t] = fmaxf(smx[t], smx[t + s]); }
    __syncthreads();
  }
  if (t == 0) { part[blockIdx.x] = smn[0]; part[256 + blockIdx.x] = smx[0]; }
}

__global__ __launch_bounds__(256)
void minmax_final(const float* __restrict__ part, float* __restrict__ scal) {
  __shared__ float smn[256], smx[256];
  int t = threadIdx.x;
  smn[t] = part[t];
  smx[t] = part[256 + t];
  __syncthreads();
  for (int s = 128; s > 0; s >>= 1) {
    if (t < s) { smn[t] = fminf(smn[t], smn[t + s]); smx[t] = fmaxf(smx[t], smx[t + s]); }
    __syncthreads();
  }
  if (t == 0) {
    float mn = smn[0], mx = smx[0];
    scal[2] = mn;
    scal[3] = 1.0f / (mx - mn);
    scal[1] = 1.0f;   // identity scale slot for non-first layers
  }
}

__global__ __launch_bounds__(256)
void normalize_k(const float* __restrict__ x, const float* __restrict__ scal,
                 float* __restrict__ y, float* __restrict__ x0) {
  float mn = scal[2], inv = scal[3];
  for (int i = blockIdx.x * 256 + threadIdx.x; i < NTOT; i += gridDim.x * 256) {
    float v = (x[i] - mn) * inv;
    y[i]  = v;
    x0[i] = v;
  }
}

// Reduce the 2048 per-conv-block partials (deterministic fixed order).
__global__ __launch_bounds__(256)
void sum_final(const float* __restrict__ part, float* __restrict__ scal) {
  __shared__ float ss[256];
  int t = threadIdx.x;
  float acc = 0.0f;
  for (int i = t; i < NBLK_CONV; i += 256) acc += part[i];
  ss[t] = acc;
  __syncthreads();
  for (int s = 128; s > 0; s >>= 1) {
    if (t < s) ss[t] += ss[t + s];
    __syncthreads();
  }
  if (t == 0) scal[0] = 1.0f / ss[0];
}

__global__ __launch_bounds__(256)
void final_scale(float* __restrict__ x, const float* __restrict__ scal) {
  float s = scal[0];
  for (int i = blockIdx.x * 256 + threadIdx.x; i < NTOT; i += gridDim.x * 256)
    x[i] *= s;
}

// ---------------------------------------------------------------------------
extern "C" void kernel_launch(void* const* d_in, const int* in_sizes, int n_in,
                              void* d_out, int out_size, void* d_ws, size_t ws_size,
                              hipStream_t stream) {
  (void)in_sizes; (void)n_in; (void)out_size; (void)ws_size;
  const float* x  = (const float*)d_in[0];
  const float* Wx = (const float*)d_in[1];
  const float* bx = (const float*)d_in[2];
  const float* Wy = (const float*)d_in[3];
  const float* by = (const float*)d_in[4];
  float* out = (float*)d_out;
  float* ws  = (float*)d_ws;

  float* ybuf = ws;                          // NTOT floats (frozen y)
  float* xalt = ws + (size_t)NTOT;           // NTOT floats (ping-pong partner)
  float* part = ws + 2 * (size_t)NTOT;       // NBLK_CONV floats
  float* scal = part + NBLK_CONV;            // [0]=1/sum [1]=1.0 [2]=min [3]=1/range

  minmax_partial<<<256, 256, 0, stream>>>(x, part);
  minmax_final<<<1, 256, 0, stream>>>(part, scal);
  normalize_k<<<4096, 256, 0, stream>>>(x, scal, ybuf, out);

  float* bufs[2] = { out, xalt };   // 50 convs (even) => final result lands in d_out
  int p = 0;
  dim3 cgrid(4, 16, 32);
  for (int it = 0; it < 10; ++it) {
    for (int l = 0; l < 5; ++l) {
      // fold previous iteration's 1/sum into the x-conv of the first layer
      const float* sp = (l == 0 && it > 0) ? (scal + 0) : (scal + 1);
      conv_layer<<<cgrid, 256, 0, stream>>>(bufs[p], ybuf,
                                            Wx + 25 * l, bx + l,
                                            Wy + 25 * l, by + l,
                                            sp, bufs[p ^ 1],
                                            part, (l == 4) ? 1 : 0);
      p ^= 1;
    }
    sum_final<<<1, 256, 0, stream>>>(part, scal);
  }
  final_scale<<<4096, 256, 0, stream>>>(out, scal);
}
